// CMAModel_9826885173458
// MI455X (gfx1250) — matrix operand order repaired and checked by reading, not compiled, not run on hardware
//
#include <hip/hip_runtime.h>
#include <hip/hip_bf16.h>
#include <math.h>

// ---------------- problem constants (match setup_inputs) ----------------
#define BB 2
#define TT 2048
#define MM 512
#define CC 1024
#define HH 16
#define DD 64
#define SS (TT + 2*MM)   // 3072

typedef __bf16 v16bf __attribute__((ext_vector_type(16)));
typedef float  v8f   __attribute__((ext_vector_type(8)));

union FragAB {
  v16bf v;
  uint4 q[2];
  unsigned short us[16];
};

__device__ __forceinline__ unsigned short f2bf(float f) {
  unsigned u = __float_as_uint(f);
  u += 0x7FFFu + ((u >> 16) & 1u);            // round-to-nearest-even
  return (unsigned short)(u >> 16);
}
__device__ __forceinline__ float bf2f(unsigned short h) {
  return __uint_as_float(((unsigned)h) << 16);
}
__device__ __forceinline__ v8f wmma_bf16(const FragAB& a, const FragAB& b, v8f c) {
  return __builtin_amdgcn_wmma_f32_16x16x32_bf16(false, a.v, false, b.v,
                                                 (short)0, c, false, false);
}
__device__ __forceinline__ float rmax16(float v) {
  v = fmaxf(v, __shfl_xor(v, 1));
  v = fmaxf(v, __shfl_xor(v, 2));
  v = fmaxf(v, __shfl_xor(v, 4));
  v = fmaxf(v, __shfl_xor(v, 8));
  return v;
}
__device__ __forceinline__ float rsum16(float v) {
  v += __shfl_xor(v, 1);
  v += __shfl_xor(v, 2);
  v += __shfl_xor(v, 4);
  v += __shfl_xor(v, 8);
  return v;
}

// ---------------- pack x / memories into bf16 ----------------
__global__ void k_pack_inputs(const float* __restrict__ x,
                              const float* __restrict__ fm,
                              const float* __restrict__ rm,
                              unsigned short* __restrict__ xbf,
                              unsigned short* __restrict__ xkv) {
  int idx = blockIdx.x * 256 + threadIdx.x;        // B*S*C = 6291456
  if (idx >= BB * SS * CC) return;
  int c = idx % CC;
  int s = (idx / CC) % SS;
  int b = idx / (CC * SS);
  float val;
  if (s < TT) {
    val = x[((long)b * TT + s) * CC + c];
    xbf[((long)b * TT + s) * CC + c] = f2bf(val);
  } else if (s < TT + MM) {
    val = fm[((long)b * MM + (s - TT)) * CC + c];
  } else {
    val = rm[((long)b * MM + (s - TT - MM)) * CC + c];
  }
  xkv[(long)idx] = f2bf(val);
}

// ---------------- transpose-convert a CxC weight to bf16 [N,K] ----------------
__global__ void k_transpose_w(const float* __restrict__ W, unsigned short* __restrict__ Wt) {
  int idx = blockIdx.x * 256 + threadIdx.x;        // 1048576
  if (idx >= CC * CC) return;
  int k = idx % CC;
  int n = idx / CC;
  Wt[(long)n * CC + k] = f2bf(W[(long)k * CC + n]);
}

// ---------------- WMMA GEMM: [Mrows,1024]bf16 x Wt[1024,1024] ----------------
// Wave computes a 32x64 output tile: 2 row-blocks share 4 B-fragments per
// K-step => 12 b128 loads per 8 WMMA (vs 10 per 4 for a 16x64 tile).
// mode 0: bf16 row-major out   mode 1: bf16 out to vT[B,H,D,S]   mode 2: f32 out
__global__ void __launch_bounds__(128) k_gemm_bf16(const unsigned short* __restrict__ A,
                                                   const unsigned short* __restrict__ Wt,
                                                   void* __restrict__ out,
                                                   int Mrows, int mode) {
  const int K = CC, N = CC;
  int wid = threadIdx.x >> 5;
  int lane = threadIdx.x & 31;
  int half = lane >> 4, l15 = lane & 15;
  int w = blockIdx.x * 4 + wid;
  int nstrips = N / 64;                            // 16
  int r0 = (w / nstrips) * 32;
  int c0 = (w % nstrips) * 64;
  if (r0 >= Mrows) return;

  v8f acc[2][4];
#pragma unroll
  for (int rt = 0; rt < 2; rt++)
#pragma unroll
    for (int ct = 0; ct < 4; ct++)
#pragma unroll
      for (int j = 0; j < 8; j++) acc[rt][ct][j] = 0.f;

  const unsigned short* arow0 = A + (long)(r0 + l15) * K;
  const unsigned short* arow1 = A + (long)(r0 + 16 + l15) * K;
  for (int k0 = 0; k0 < K; k0 += 32) {
    FragAB a[2];
    a[0].q[0] = *(const uint4*)(arow0 + k0 + 8 * half);
    a[0].q[1] = *(const uint4*)(arow0 + k0 + 16 + 8 * half);
    a[1].q[0] = *(const uint4*)(arow1 + k0 + 8 * half);
    a[1].q[1] = *(const uint4*)(arow1 + k0 + 16 + 8 * half);
#pragma unroll
    for (int ct = 0; ct < 4; ct++) {
      FragAB bfr;
      const unsigned short* wrow = Wt + (long)(c0 + 16 * ct + l15) * K + k0 + 8 * half;
      bfr.q[0] = *(const uint4*)(wrow);
      bfr.q[1] = *(const uint4*)(wrow + 16);
#pragma unroll
      for (int rt = 0; rt < 2; rt++)
        acc[rt][ct] = wmma_bf16(a[rt], bfr, acc[rt][ct]);
    }
  }

  if (mode == 0) {
    unsigned short* O = (unsigned short*)out;
#pragma unroll
    for (int rt = 0; rt < 2; rt++)
#pragma unroll
      for (int ct = 0; ct < 4; ct++)
#pragma unroll
        for (int j = 0; j < 8; j++) {
          int row = r0 + 16 * rt + j + 8 * half;
          int col = c0 + 16 * ct + l15;
          O[(long)row * N + col] = f2bf(acc[rt][ct][j]);
        }
  } else if (mode == 2) {
    float* O = (float*)out;
#pragma unroll
    for (int rt = 0; rt < 2; rt++)
#pragma unroll
      for (int ct = 0; ct < 4; ct++)
#pragma unroll
        for (int j = 0; j < 8; j++) {
          int row = r0 + 16 * rt + j + 8 * half;
          int col = c0 + 16 * ct + l15;
          O[(long)row * N + col] = acc[rt][ct][j];
        }
  } else {                                         // V transposed per head
    unsigned short* O = (unsigned short*)out;
#pragma unroll
    for (int rt = 0; rt < 2; rt++)
#pragma unroll
      for (int ct = 0; ct < 4; ct++)
#pragma unroll
        for (int j = 0; j < 8; j++) {
          int row = r0 + 16 * rt + j + 8 * half;   // = b*S + s
          int col = c0 + 16 * ct + l15;            // = h*64 + d
          int b = row / SS, s = row - b * SS;
          int h = col >> 6, d = col & 63;
          O[(((long)b * HH + h) * DD + d) * SS + s] = f2bf(acc[rt][ct][j]);
        }
  }
}

// ---------------- gate logits / sigmoid / L1 reg loss ----------------
__global__ void k_gate(const unsigned short* __restrict__ qbf,
                       const float* __restrict__ gateW,
                       const float* __restrict__ gateB,
                       float* __restrict__ g, float* __restrict__ loss) {
  __shared__ float red[256];
  int idx = blockIdx.x * 256 + threadIdx.x;        // B*T*H = 65536
  int h = idx % HH;
  int t = (idx / HH) % TT;
  int b = idx / (HH * TT);
  const unsigned short* qr = qbf + ((long)b * TT + t) * CC;
  float acc = 0.f;
  for (int c = 0; c < CC; c++) acc += bf2f(qr[c]) * gateW[c * HH + h];
  acc += gateB[h];
  float gv = 1.f / (1.f + __expf(-acc));
  g[((long)b * HH + h) * TT + t] = gv;

  red[threadIdx.x] = gv;
  __syncthreads();
  for (int s = 128; s > 0; s >>= 1) {
    if (threadIdx.x < s) red[threadIdx.x] += red[threadIdx.x + s];
    __syncthreads();
  }
  if (threadIdx.x == 0)
    atomicAdd(loss, red[0] * (0.01f / (float)(BB * TT * HH)));
}

__global__ void k_zero(float* p) { if (threadIdx.x == 0) *p = 0.f; }

// ---------------- flash attention with gated local/memory split ----------------
__global__ void __launch_bounds__(128) k_attn(const unsigned short* __restrict__ qbf,
                                              const unsigned short* __restrict__ kbf,
                                              const unsigned short* __restrict__ vT,
                                              const float* __restrict__ g,
                                              unsigned short* __restrict__ Ybf) {
  int wid = threadIdx.x >> 5, lane = threadIdx.x & 31;
  int half = lane >> 4, l15 = lane & 15;
  int blk = blockIdx.x;                            // B*H*(T/64) = 1024
  int tstrip = blk % (TT / 64);
  int bh = blk / (TT / 64);
  int h = bh % HH, b = bh / HH;
  int t0 = tstrip * 64 + wid * 16;

  __shared__ unsigned short pT[4][16 * 32];
  unsigned short* lds = &pT[wid][0];

  FragAB qa[2];                                    // Q 16x64, resident
  {
    const unsigned short* qrow = qbf + ((long)b * TT + t0 + l15) * CC + h * DD;
    qa[0].q[0] = *(const uint4*)(qrow + 8 * half);
    qa[0].q[1] = *(const uint4*)(qrow + 16 + 8 * half);
    qa[1].q[0] = *(const uint4*)(qrow + 32 + 8 * half);
    qa[1].q[1] = *(const uint4*)(qrow + 48 + 8 * half);
  }

  float m[8], l[8];
  v8f oL[4], oM[4];
#pragma unroll
  for (int j = 0; j < 8; j++) { m[j] = -1.0e30f; l[j] = 0.f; }
#pragma unroll
  for (int dt = 0; dt < 4; dt++)
#pragma unroll
    for (int j = 0; j < 8; j++) { oL[dt][j] = 0.f; oM[dt][j] = 0.f; }

  auto tile = [&](int s0, bool domask, v8f (&oAcc)[4]) {
    // ---- scores: Q (16x64) x K^T (64x32) via 4 WMMA ----
    v8f sc[2];
#pragma unroll
    for (int ct = 0; ct < 2; ct++) {
      v8f a;
#pragma unroll
      for (int j = 0; j < 8; j++) a[j] = 0.f;
#pragma unroll
      for (int ds = 0; ds < 2; ds++) {
        FragAB kb;
        const unsigned short* kr =
            kbf + ((long)b * SS + s0 + 16 * ct + l15) * CC + h * DD + 32 * ds + 8 * half;
        kb.q[0] = *(const uint4*)kr;
        kb.q[1] = *(const uint4*)(kr + 16);
        a = wmma_bf16(qa[ds], kb, a);
      }
      sc[ct] = a;
    }
    // ---- scale + causal mask ----
#pragma unroll
    for (int ct = 0; ct < 2; ct++)
#pragma unroll
      for (int j = 0; j < 8; j++) {
        float s = sc[ct][j] * 0.125f;              // 1/sqrt(64)
        if (domask) {
          int col = s0 + 16 * ct + l15;
          int row = t0 + j + 8 * half;
          if (col > row) s = -1.0e30f;
        }
        sc[ct][j] = s;
      }
    // ---- streaming softmax stats + P to LDS (A-fragment relayout) ----
#pragma unroll
    for (int j = 0; j < 8; j++) {
      float rm = rmax16(fmaxf(sc[0][j], sc[1][j]));
      float mn = fmaxf(m[j], rm);
      float alpha = __expf(m[j] - mn);
      m[j] = mn;
      float p0 = (sc[0][j] > -1.0e29f) ? __expf(sc[0][j] - mn) : 0.f;
      float p1 = (sc[1][j] > -1.0e29f) ? __expf(sc[1][j] - mn) : 0.f;
      l[j] = l[j] * alpha + rsum16(p0 + p1);
#pragma unroll
      for (int dt = 0; dt < 4; dt++) { oL[dt][j] *= alpha; oM[dt][j] *= alpha; }
      int row = j + 8 * half;
      lds[row * 32 + l15] = f2bf(p0);
      lds[row * 32 + 16 + l15] = f2bf(p1);
    }
    asm volatile("" ::: "memory");                 // same-wave LDS in-order
    FragAB pa;
    pa.q[0] = *(const uint4*)(lds + l15 * 32 + 8 * half);
    pa.q[1] = *(const uint4*)(lds + l15 * 32 + 16 + 8 * half);
    asm volatile("" ::: "memory");
    // ---- O += P (16x32) x V (32x64) via 4 WMMA ----
#pragma unroll
    for (int dt = 0; dt < 4; dt++) {
      FragAB vb;
      const unsigned short* vr =
          vT + (((long)b * HH + h) * DD + 16 * dt + l15) * SS + s0 + 8 * half;
      vb.q[0] = *(const uint4*)vr;
      vb.q[1] = *(const uint4*)(vr + 16);
      oAcc[dt] = wmma_bf16(pa, vb, oAcc[dt]);
    }
  };

  for (int s0 = 0; s0 < t0 + 16; s0 += 32) tile(s0, true, oL);   // local causal keys
  for (int s0 = TT; s0 < SS; s0 += 32) tile(s0, false, oM);      // memory keys

  // ---- merge: Y = (O_local + g*O_mem)/l, store bf16 ----
#pragma unroll
  for (int j = 0; j < 8; j++) {
    int row = t0 + j + 8 * half;
    float inv = 1.f / l[j];
    float gv = g[((long)b * HH + h) * TT + row];
#pragma unroll
    for (int dt = 0; dt < 4; dt++) {
      float y = (oL[dt][j] + gv * oM[dt][j]) * inv;
      Ybf[((long)b * TT + row) * CC + h * DD + 16 * dt + l15] = f2bf(y);
    }
  }
}

// ---------------- host orchestration ----------------
extern "C" void kernel_launch(void* const* d_in, const int* in_sizes, int n_in,
                              void* d_out, int out_size, void* d_ws, size_t ws_size,
                              hipStream_t stream) {
  (void)in_sizes; (void)n_in; (void)out_size; (void)ws_size;
  const float* x   = (const float*)d_in[0];
  const float* fm  = (const float*)d_in[1];
  const float* rm  = (const float*)d_in[2];
  const float* Wq  = (const float*)d_in[3];
  const float* Wk  = (const float*)d_in[4];
  const float* Wv  = (const float*)d_in[5];
  const float* Wo  = (const float*)d_in[6];
  const float* gW  = (const float*)d_in[7];
  const float* gB  = (const float*)d_in[8];

  char* ws = (char*)d_ws;
  size_t off = 0;
  unsigned short* xbf = (unsigned short*)(ws + off); off += (size_t)BB*TT*CC*2;   // 8 MB
  unsigned short* xkv = (unsigned short*)(ws + off); off += (size_t)BB*SS*CC*2;   // 12 MB
  unsigned short* Wqt = (unsigned short*)(ws + off); off += (size_t)CC*CC*2;
  unsigned short* Wkt = (unsigned short*)(ws + off); off += (size_t)CC*CC*2;
  unsigned short* Wvt = (unsigned short*)(ws + off); off += (size_t)CC*CC*2;
  unsigned short* Wot = (unsigned short*)(ws + off); off += (size_t)CC*CC*2;
  unsigned short* qbf = (unsigned short*)(ws + off); off += (size_t)BB*TT*CC*2;
  unsigned short* kbf = (unsigned short*)(ws + off); off += (size_t)BB*SS*CC*2;
  unsigned short* vT  = (unsigned short*)(ws + off); off += (size_t)BB*SS*CC*2;
  unsigned short* Ybf = (unsigned short*)(ws + off); off += (size_t)BB*TT*CC*2;
  float*          gArr= (float*)(ws + off);          off += (size_t)BB*HH*TT*4;

  float* outF = (float*)d_out;
  float* loss = outF + (size_t)BB * TT * CC;

  k_zero<<<1, 32, 0, stream>>>(loss);

  k_pack_inputs<<<(BB*SS*CC + 255)/256, 256, 0, stream>>>(x, fm, rm, xbf, xkv);

  k_transpose_w<<<(CC*CC + 255)/256, 256, 0, stream>>>(Wq, Wqt);
  k_transpose_w<<<(CC*CC + 255)/256, 256, 0, stream>>>(Wk, Wkt);
  k_transpose_w<<<(CC*CC + 255)/256, 256, 0, stream>>>(Wv, Wvt);
  k_transpose_w<<<(CC*CC + 255)/256, 256, 0, stream>>>(Wo, Wot);

  // Q = x @ Wq  -> qbf (bf16)    waves = (4096/32)*(1024/64) = 2048 -> 512 blocks
  k_gemm_bf16<<<(BB*TT/32)*(CC/64)/4, 128, 0, stream>>>(xbf, Wqt, qbf, BB*TT, 0);
  // K = [x;fm;rm] @ Wk -> kbf    waves = (6144/32)*16 = 3072 -> 768 blocks
  k_gemm_bf16<<<(BB*SS/32)*(CC/64)/4, 128, 0, stream>>>(xkv, Wkt, kbf, BB*SS, 0);
  // V = [x;fm;rm] @ Wv -> vT (bf16, per-head transposed [B,H,D,S])
  k_gemm_bf16<<<(BB*SS/32)*(CC/64)/4, 128, 0, stream>>>(xkv, Wvt, vT, BB*SS, 1);

  // gate: g = sigmoid(q @ gate_W + b), plus L1 reg loss
  k_gate<<<(BB*TT*HH)/256, 256, 0, stream>>>(qbf, gW, gB, gArr, loss);

  // attention with gated local/memory merge -> Ybf (bf16)
  k_attn<<<BB*HH*(TT/64), 128, 0, stream>>>(qbf, kbf, vT, gArr, Ybf);

  // out = Y @ Wo -> f32 d_out
  k_gemm_bf16<<<(BB*TT/32)*(CC/64)/4, 128, 0, stream>>>(Ybf, Wot, outF, BB*TT, 2);
}
